// Discriminator_455266534113
// MI455X (gfx1250) — compile-verified, matrix operand order checked by reading, not run on hardware
//
#include <hip/hip_runtime.h>

typedef __attribute__((ext_vector_type(8)))  _Float16 v8h;
typedef __attribute__((ext_vector_type(16))) _Float16 v16h;
typedef __attribute__((ext_vector_type(8)))  float    v8f;

#define DEVINL __device__ __forceinline__

// Padded leading dims: +8 halves (16B) per row -> row strides 80/144/272 B.
// gcd(stride/4, 64) == 4 in every case, so 16 lanes touching 16 consecutive
// rows hit 16 distinct LDS bank groups: conflict-free ds_load_b128 / stores,
// while rows stay 16B-aligned so fragment chunks remain contiguous b128s.
#define LDP32  40
#define LDP64  72
#define LDP128 136
#define ARR    (128 * LDP128)   // elements per LDS matrix (17408)

// Load a 16x32 f16 fragment (CDNA5 WMMA A-layout, ISA 7.12.2) from LDS,
// row-major with leading dim `ld`. Lane = hi*16 + l holds element e =
// (row0+l, k0 + (e<8?0:16) + hi*8 + (e&7)): two contiguous 16B chunks.
// The same layout serves the B operand when B is stored transposed.
DEVINL v16h frag_ld(const _Float16* __restrict__ base, int row0, int k0, int ld, int lane) {
  const int l = lane & 15, hi = lane >> 4;
  const _Float16* p = base + (row0 + l) * ld + k0 + hi * 8;
  v8h lo = *reinterpret_cast<const v8h*>(p);
  v8h hv = *reinterpret_cast<const v8h*>(p + 16);
  return __builtin_shufflevector(lo, hv, 0,1,2,3,4,5,6,7,8,9,10,11,12,13,14,15);
}

// One 16-row output stripe x NT column tiles, k-outer so the A fragment is
// loaded once per k-step and reused by all NT WMMAs (SRC0 reuse).
template<int K, int NT>
DEVINL void gemm_stripe(const _Float16* __restrict__ sA, int lda,
                        const _Float16* __restrict__ sBT, int ldb,
                        int rt, int lane, v8f* acc) {
#pragma unroll
  for (int kk = 0; kk < K; kk += 32) {
    v16h a = frag_ld(sA, rt, kk, lda, lane);
#pragma unroll
    for (int i = 0; i < NT; ++i) {
      v16h b = frag_ld(sBT, i * 16, kk, ldb, lane);
      acc[i] = __builtin_amdgcn_wmma_f32_16x16x32_f16(false, a, false, b, (short)0,
                                                      acc[i], false, false);
    }
  }
}

DEVINL float sigmoidf_(float x) { return 1.0f / (1.0f + expf(-x)); }

__global__ __launch_bounds__(256)
void rgcn_fused(const float* __restrict__ A,   const float* __restrict__ X,
                const float* __restrict__ Wl1, const float* __restrict__ bl1,
                const float* __restrict__ Wf1, const float* __restrict__ bf1,
                const float* __restrict__ Wl2, const float* __restrict__ bl2,
                const float* __restrict__ Wf2, const float* __restrict__ bf2,
                const float* __restrict__ Wi,  const float* __restrict__ bi,
                const float* __restrict__ Wj,  const float* __restrict__ bj,
                const float* __restrict__ W1,  const float* __restrict__ b1,
                const float* __restrict__ W2,  const float* __restrict__ b2,
                float* __restrict__ out) {
  extern __shared__ _Float16 smem[];
  _Float16* sAdj = smem;            // [128][LDP128] current-relation adjacency
  _Float16* sBT  = smem + ARR;      // [H][LDPK]  transposed right operand
  _Float16* sHT  = smem + 2 * ARR;  // [H][LDP128] transposed h_r; later i row-major
  _Float16* sAct = smem + 3 * ARR;  // activations: X (LDP32) | x1 (LDP64) | x2 (LDP128)
  __shared__ float sG[128];
  __shared__ float sH[128];

  const int b    = blockIdx.x;
  const int tid  = threadIdx.x;
  const int lane = tid & 31;
  const int wave = tid >> 5;
  const int l    = lane & 15;
  const int hi   = lane >> 4;
  const int rt   = wave * 16;       // each wave owns one 16-row output stripe

  const size_t adjBase = (size_t)b * (128 * 128 * 5);

  // ---- stage X_b (f32 -> f16), [128][LDP32] ----
  for (int idx = tid; idx < 128 * 32; idx += 256) {
    int m = idx >> 5, n = idx & 31;
    sAct[m * LDP32 + n] = (_Float16)X[(size_t)b * (128 * 32) + idx];
  }

  // ---- layer 1: feature branch seeds the accumulators ----
  for (int idx = tid; idx < 32 * 64; idx += 256) {      // Wf1 [32][64] -> BT [64][LDP32]
    int k = idx >> 6, h = idx & 63;
    sBT[h * LDP32 + k] = (_Float16)Wf1[idx];
  }
  __syncthreads();

  v8f acc1[4];
#pragma unroll
  for (int i = 0; i < 4; ++i) acc1[i] = v8f{};
  gemm_stripe<32, 4>(sAct, LDP32, sBT, LDP32, rt, lane, acc1);
#pragma unroll
  for (int i = 0; i < 4; ++i) {
    float bb = bf1[i * 16 + l];
#pragma unroll
    for (int e = 0; e < 8; ++e) {
      float v = acc1[i][e] + bb;
      acc1[i][e] = v > 0.0f ? v : 0.0f;
    }
  }
  __syncthreads();

  // ---- layer 1: per-relation projection + adjacency aggregation ----
  for (int r = 0; r < 5; ++r) {
    const float* W = Wl1 + r * (32 * 64);
    for (int idx = tid; idx < 32 * 64; idx += 256) {
      int k = idx >> 6, h = idx & 63;
      sBT[h * LDP32 + k] = (_Float16)W[idx];
    }
    const float* Ab = A + adjBase + r;                  // adj[b][r][m][n] = A[b][m][n][r]
    for (int idx = tid; idx < 128 * 128; idx += 256) {
      int m = idx >> 7, n = idx & 127;
      sAdj[m * LDP128 + n] = (_Float16)Ab[(size_t)idx * 5];
    }
    __syncthreads();

    v8f hacc[4];
#pragma unroll
    for (int i = 0; i < 4; ++i) hacc[i] = v8f{};
    gemm_stripe<32, 4>(sAct, LDP32, sBT, LDP32, rt, lane, hacc);
    const float* blr = bl1 + r * 64;
#pragma unroll
    for (int i = 0; i < 4; ++i) {
      int col = i * 16 + l;
      float bb = blr[col];
#pragma unroll
      for (int e = 0; e < 8; ++e) {
        float v = hacc[i][e] + bb;
        v = v > 0.0f ? v : 0.0f;
        sHT[col * LDP128 + (rt + hi * 8 + e)] = (_Float16)v; // transposed [h][n]
      }
    }
    __syncthreads();

    gemm_stripe<128, 4>(sAdj, LDP128, sHT, LDP128, rt, lane, acc1);
    __syncthreads();
  }

  // x1 = tanh(acc1) -> sAct+5120 as [128][LDP64]
  _Float16* sX1 = sAct + 128 * LDP32;
#pragma unroll
  for (int i = 0; i < 4; ++i) {
    int col = i * 16 + l;
#pragma unroll
    for (int e = 0; e < 8; ++e)
      sX1[(rt + hi * 8 + e) * LDP64 + col] = (_Float16)tanhf(acc1[i][e]);
  }
  __syncthreads();

  // ---- layer 2: feature branch seed ----
  for (int idx = tid; idx < 64 * 128; idx += 256) {     // Wf2 [64][128] -> BT [128][LDP64]
    int k = idx >> 7, h = idx & 127;
    sBT[h * LDP64 + k] = (_Float16)Wf2[idx];
  }
  __syncthreads();

  v8f acc2[8];
#pragma unroll
  for (int i = 0; i < 8; ++i) acc2[i] = v8f{};
  gemm_stripe<64, 8>(sX1, LDP64, sBT, LDP64, rt, lane, acc2);
#pragma unroll
  for (int i = 0; i < 8; ++i) {
    float bb = bf2[i * 16 + l];
#pragma unroll
    for (int e = 0; e < 8; ++e) {
      float v = acc2[i][e] + bb;
      acc2[i][e] = v > 0.0f ? v : 0.0f;
    }
  }
  __syncthreads();

  // ---- layer 2: relations ----
  for (int r = 0; r < 5; ++r) {
    const float* W = Wl2 + r * (64 * 128);
    for (int idx = tid; idx < 64 * 128; idx += 256) {
      int k = idx >> 7, h = idx & 127;
      sBT[h * LDP64 + k] = (_Float16)W[idx];
    }
    const float* Ab = A + adjBase + r;
    for (int idx = tid; idx < 128 * 128; idx += 256) {
      int m = idx >> 7, n = idx & 127;
      sAdj[m * LDP128 + n] = (_Float16)Ab[(size_t)idx * 5]; // 2nd pass: L2-resident
    }
    __syncthreads();

    v8f hacc[8];
#pragma unroll
    for (int i = 0; i < 8; ++i) hacc[i] = v8f{};
    gemm_stripe<64, 8>(sX1, LDP64, sBT, LDP64, rt, lane, hacc);
    const float* blr = bl2 + r * 128;
#pragma unroll
    for (int i = 0; i < 8; ++i) {
      int col = i * 16 + l;
      float bb = blr[col];
#pragma unroll
      for (int e = 0; e < 8; ++e) {
        float v = hacc[i][e] + bb;
        v = v > 0.0f ? v : 0.0f;
        sHT[col * LDP128 + (rt + hi * 8 + e)] = (_Float16)v;
      }
    }
    __syncthreads();

    gemm_stripe<128, 8>(sAdj, LDP128, sHT, LDP128, rt, lane, acc2);
    __syncthreads();
  }

  // x2 = tanh(acc2) -> sAct as [128][LDP128] (X/x1 are dead)
#pragma unroll
  for (int i = 0; i < 8; ++i) {
    int col = i * 16 + l;
#pragma unroll
    for (int e = 0; e < 8; ++e)
      sAct[(rt + hi * 8 + e) * LDP128 + col] = (_Float16)tanhf(acc2[i][e]);
  }
  __syncthreads();

  // ---- gated aggregation: i = sigmoid(x2 @ Wi + bi) -> sHT row-major ----
  for (int idx = tid; idx < 128 * 128; idx += 256) {
    int k = idx >> 7, h = idx & 127;
    sBT[h * LDP128 + k] = (_Float16)Wi[idx];
  }
  __syncthreads();

  {
    v8f ai[8];
#pragma unroll
    for (int i = 0; i < 8; ++i) ai[i] = v8f{};
    gemm_stripe<128, 8>(sAct, LDP128, sBT, LDP128, rt, lane, ai);
#pragma unroll
    for (int i = 0; i < 8; ++i) {
      int col = i * 16 + l;
      float bb = bi[col];
#pragma unroll
      for (int e = 0; e < 8; ++e)
        sHT[(rt + hi * 8 + e) * LDP128 + col] = (_Float16)sigmoidf_(ai[i][e] + bb);
    }
  }
  if (tid < 128) sG[tid] = 0.0f;
  __syncthreads();

  // ---- j = tanh(i @ Wj + bj) fused with g = sum_n i*j (ds_add_f32/column) ----
  for (int idx = tid; idx < 128 * 128; idx += 256) {
    int k = idx >> 7, h = idx & 127;
    sBT[h * LDP128 + k] = (_Float16)Wj[idx];
  }
  __syncthreads();

  {
    v8f aj[8];
#pragma unroll
    for (int i = 0; i < 8; ++i) aj[i] = v8f{};
    gemm_stripe<128, 8>(sHT, LDP128, sBT, LDP128, rt, lane, aj);
#pragma unroll
    for (int i = 0; i < 8; ++i) {
      int col = i * 16 + l;
      float bb = bj[col];
      float part = 0.0f;
#pragma unroll
      for (int e = 0; e < 8; ++e) {
        float jv = tanhf(aj[i][e] + bb);
        part += jv * (float)sHT[(rt + hi * 8 + e) * LDP128 + col];   // i value
      }
      atomicAdd(&sG[col], part);   // all 8 elements of a lane share this column
    }
  }
  __syncthreads();

  // ---- head: g = tanh(sum); h = tanh(g@W1+b1); out = tanh(h@W2+b2) ----
  if (tid < 128) sH[tid] = tanhf(sG[tid]);
  __syncthreads();
  if (tid < 128) {
    float a = b1[tid];
    for (int f = 0; f < 128; ++f) a += sH[f] * W1[f * 128 + tid];
    sG[tid] = tanhf(a);
  }
  __syncthreads();
  if (tid == 0) {
    float s = b2[0];
    for (int f = 0; f < 128; ++f) s += sG[f] * W2[f];
    out[b] = tanhf(s);
  }
}

extern "C" void kernel_launch(void* const* d_in, const int* in_sizes, int n_in,
                              void* d_out, int out_size, void* d_ws, size_t ws_size,
                              hipStream_t stream) {
  const float* A   = (const float*)d_in[0];
  const float* X   = (const float*)d_in[1];
  const float* Wl1 = (const float*)d_in[2];
  const float* bl1 = (const float*)d_in[3];
  const float* Wf1 = (const float*)d_in[4];
  const float* bf1 = (const float*)d_in[5];
  const float* Wl2 = (const float*)d_in[6];
  const float* bl2 = (const float*)d_in[7];
  const float* Wf2 = (const float*)d_in[8];
  const float* bf2 = (const float*)d_in[9];
  const float* Wi  = (const float*)d_in[10];
  const float* bi  = (const float*)d_in[11];
  const float* Wj  = (const float*)d_in[12];
  const float* bj  = (const float*)d_in[13];
  const float* W1  = (const float*)d_in[14];
  const float* b1  = (const float*)d_in[15];
  const float* W2  = (const float*)d_in[16];
  const float* b2  = (const float*)d_in[17];
  float* out = (float*)d_out;

  const int B = out_size;                        // output is [B,1] fp32
  const size_t smem = 4u * (size_t)ARR * 2u;     // 4 padded 128x136 f16 mats = 136 KiB
  (void)hipFuncSetAttribute((const void*)rgcn_fused,
                            hipFuncAttributeMaxDynamicSharedMemorySize, (int)smem);
  rgcn_fused<<<B, 256, smem, stream>>>(A, X, Wl1, bl1, Wf1, bf1, Wl2, bl2, Wf2, bf2,
                                       Wi, bi, Wj, bj, W1, b1, W2, b2, out);
}
